// RGLRU_10505490006121
// MI455X (gfx1250) — matrix operand order repaired and checked by reading, not compiled, not run on hardware
//
#include <hip/hip_runtime.h>
#include <math.h>

// ---------------- CDNA5 (gfx1250, wave32) RG-LRU forward ----------------
// GEMMs: V_WMMA_F32_16X16X32_BF16 with bf16x3 split emulation. The f32->bf16
// hi/lo split is done ONCE in global memory (pre-pass / producer epilogues),
// so the GEMM hot loop is pure {async global->LDS copy, ds_load_b128, wmma}
// with zero conversion VALU.

typedef __attribute__((ext_vector_type(16))) __bf16 v16bf;
typedef __attribute__((ext_vector_type(8)))  __bf16 v8bf;
typedef __attribute__((ext_vector_type(4)))  __bf16 v4bf;
typedef __attribute__((ext_vector_type(8)))  float  v8f;
typedef __attribute__((ext_vector_type(4)))  float  f4;
typedef __attribute__((ext_vector_type(4)))  int    i4;

#define DIMC  1024
#define HIDC  2048
#define TC    2048
#define BC    4
#define NROW  (BC * TC)        // 8192
#define GI_LD (2 * HIDC)       // 4096

#define TILE_M 128
#define TILE_N 128
#define TILE_K 32
#define LDSB   40              // padded LDS row stride in bf16 (80B): conflict-free

#if __has_builtin(__builtin_amdgcn_global_load_async_to_lds_b128) && \
    __has_builtin(__builtin_amdgcn_s_wait_asynccnt)
#define HAVE_ASYNC 1
#else
#define HAVE_ASYNC 0
#endif

#if HAVE_ASYNC
typedef __attribute__((address_space(1))) i4 g_i4;   // global int4
typedef __attribute__((address_space(3))) i4 l_i4;   // LDS int4
#endif

// 16-byte global -> LDS copy: async DMA (ASYNCcnt) when available.
__device__ __forceinline__ void cp16_g2l(const void* g, void* l)
{
#if HAVE_ASYNC
    __builtin_amdgcn_global_load_async_to_lds_b128((g_i4*)g, (l_i4*)l, 0, 0);
#else
    *(i4*)l = *(const i4*)g;
#endif
}
__device__ __forceinline__ void cp_wait_all()
{
#if HAVE_ASYNC
    __builtin_amdgcn_s_wait_asynccnt(0);
#endif
}

// ---------------------------------------------------------------------------
// GEMM: C[M,N] = (Ahi+Alo)[M,Kd] * (Whi+Wlo)[N,Kd]^T (+bias), bf16 planes in,
// f32 out. 256 threads = 8 waves (2x4), 128x128 tile, BK=32, double-buffered.
// ---------------------------------------------------------------------------
__global__ __launch_bounds__(256)
void rg_gemm_bf16x3(const __bf16* __restrict__ Ahi, const __bf16* __restrict__ Alo,
                    const __bf16* __restrict__ Whi, const __bf16* __restrict__ Wlo,
                    float* __restrict__ C, const float* __restrict__ bias,
                    int Kd, int lda, int ldc)
{
    __shared__ __bf16 sA[2][2][TILE_M * LDSB];   // [buf][hi/lo][...]
    __shared__ __bf16 sB[2][2][TILE_N * LDSB];

    const int tid  = threadIdx.x;
    const int lane = tid & 31;
    const int wave = tid >> 5;
    const int lh   = lane >> 4;
    const int l15  = lane & 15;
    const int m0w  = (wave >> 2) * 64;
    const int n0w  = (wave & 3) * 32;
    const long rowBase = (long)blockIdx.y * TILE_M;
    const long colBase = (long)blockIdx.x * TILE_N;
    const int nk = Kd / TILE_K;

    // Issue one K-tile's copies: per thread 2 chunks/row-plane (16B of bf16).
    auto issue = [&](int kt, int buf) {
        const long ka = (long)kt * TILE_K;
#pragma unroll
        for (int p = 0; p < 2; ++p) {
            const int q = p * 256 + tid;
            const int r = q >> 2, c = q & 3;            // 4x 8-bf16 chunks per row
            const long goa = (rowBase + r) * (long)lda + ka + c * 8;
            const long gow = (colBase + r) * (long)Kd  + ka + c * 8;
            const int  lo  = r * LDSB + c * 8;
            cp16_g2l(Ahi + goa, &sA[buf][0][lo]);
            cp16_g2l(Alo + goa, &sA[buf][1][lo]);
            cp16_g2l(Whi + gow, &sB[buf][0][lo]);
            cp16_g2l(Wlo + gow, &sB[buf][1][lo]);
        }
    };

    v8f acc[4][2];
#pragma unroll
    for (int i = 0; i < 4; ++i)
#pragma unroll
        for (int j = 0; j < 2; ++j)
#pragma unroll
            for (int r = 0; r < 8; ++r) acc[i][j][r] = 0.0f;

    issue(0, 0);
    cp_wait_all();
    __syncthreads();

    for (int kt = 0; kt < nk; ++kt) {
        const int cur = kt & 1;
        if (kt + 1 < nk) issue(kt + 1, cur ^ 1);

        // Fragments straight from LDS (documented CDNA5 layouts):
        // A (16-bit, 16x32): elems 0..7 -> K=half*8+e ; 8..15 -> K=16+half*8+e
        // B (16-bit, 32x16): elems 0..7 -> K=half*8+e ; 8..15 -> K=16+half*8+e
        v16bf afr[2][4], bfr[2][2];
#pragma unroll
        for (int pl = 0; pl < 2; ++pl) {
#pragma unroll
            for (int mt = 0; mt < 4; ++mt) {
                const int row = m0w + mt * 16 + l15;
                v8bf a0 = *(const v8bf*)(&sA[cur][pl][row * LDSB + lh * 8]);
                v8bf a1 = *(const v8bf*)(&sA[cur][pl][row * LDSB + 16 + lh * 8]);
                afr[pl][mt] = __builtin_shufflevector(a0, a1,
                    0,1,2,3,4,5,6,7,8,9,10,11,12,13,14,15);
            }
#pragma unroll
            for (int nt = 0; nt < 2; ++nt) {
                const int row = n0w + nt * 16 + l15;
                v8bf b0 = *(const v8bf*)(&sB[cur][pl][row * LDSB + lh * 8]);
                v8bf b1 = *(const v8bf*)(&sB[cur][pl][row * LDSB + 16 + lh * 8]);
                bfr[pl][nt] = __builtin_shufflevector(b0, b1,
                    0,1,2,3,4,5,6,7,8,9,10,11,12,13,14,15);
            }
        }

        // 3 split terms (hi*hi, hi*lo, lo*hi), interleaved across the 8 acc
        // tiles so dependent WMMAs are >=7 apart (no hazard stalls).
#pragma unroll
        for (int term = 0; term < 3; ++term) {
            const int pa = (term == 2) ? 1 : 0;
            const int pb = (term == 1) ? 1 : 0;
#pragma unroll
            for (int mt = 0; mt < 4; ++mt)
#pragma unroll
                for (int nt = 0; nt < 2; ++nt)
                    acc[mt][nt] = __builtin_amdgcn_wmma_f32_16x16x32_bf16(
                        false, afr[pa][mt], false, bfr[pb][nt],
                        (short)0, acc[mt][nt], false, false);
        }

        if (kt + 1 < nk) cp_wait_all();
        __syncthreads();
    }

    // C layout: VGPR r holds M = r (lanes 0-15) / 8+r (lanes 16-31)
#pragma unroll
    for (int mt = 0; mt < 4; ++mt)
#pragma unroll
        for (int nt = 0; nt < 2; ++nt) {
            const long col = colBase + n0w + nt * 16 + l15;
            const float bv = bias ? bias[col] : 0.0f;
#pragma unroll
            for (int r = 0; r < 8; ++r) {
                const long row = rowBase + m0w + mt * 16 + lh * 8 + r;
                C[row * ldc + col] = acc[mt][nt][r] + bv;
            }
        }
}

// ---------------------------------------------------------------------------
// f32 -> (bf16 hi, bf16 lo) plane split; one float4 per thread.
// ---------------------------------------------------------------------------
__global__ __launch_bounds__(256)
void rg_split(const float* __restrict__ in, __bf16* __restrict__ hi,
              __bf16* __restrict__ lo)
{
    const long i = (long)blockIdx.x * blockDim.x + threadIdx.x;  // float4 index
    const f4 v = ((const f4*)in)[i];
    v4bf h, l;
#pragma unroll
    for (int e = 0; e < 4; ++e) {
        const __bf16 hh = (__bf16)v[e];
        h[e] = hh;
        l[e] = (__bf16)(v[e] - (float)hh);
    }
    ((v4bf*)hi)[i] = h;
    ((v4bf*)lo)[i] = l;
}

// ---------------------------------------------------------------------------
// Causal depthwise conv; emits f32 value + bf16 hi/lo planes for GEMM2.
// ---------------------------------------------------------------------------
__global__ __launch_bounds__(256)
void rg_dwconv(const float* __restrict__ gi, const float* __restrict__ cw,
               const float* __restrict__ cb, float* __restrict__ xf,
               __bf16* __restrict__ xhi, __bf16* __restrict__ xlo)
{
    const long idx = (long)blockIdx.x * blockDim.x + threadIdx.x;
    const int  h  = (int)(idx & (HIDC - 1));
    const long bt = idx >> 11;
    const int  t  = (int)(bt & (TC - 1));
    float s = cb[h];
#pragma unroll
    for (int j = 0; j < 4; ++j) {
        const int tt = t - 3 + j;
        if (tt >= 0)
            s += cw[h * 4 + j] * gi[(bt - 3 + j) * GI_LD + HIDC + h];
    }
    xf[idx] = s;
    const __bf16 hh = (__bf16)s;
    xhi[idx] = hh;
    xlo[idx] = (__bf16)(s - (float)hh);
}

// ---------------------------------------------------------------------------
// Gate math, in place over g: forget->alpha, inp->bterm
// ---------------------------------------------------------------------------
__global__ __launch_bounds__(256)
void rg_gates(float* __restrict__ g, const float* __restrict__ xc,
              const float* __restrict__ fb)
{
    const long idx = (long)blockIdx.x * blockDim.x + threadIdx.x;
    const int  h  = (int)(idx & (HIDC - 1));
    const long bt = idx >> 11;
    const float forget = g[bt * GI_LD + h];
    const float inp    = g[bt * GI_LD + HIDC + h];
    const float x      = xc[idx];
    const float sp   = log1pf(expf(fb[h]));
    const float sigf = 1.0f / (1.0f + expf(-forget));
    const float alpha = expf(-8.0f * sp * sigf);
    const float beta  = sqrtf(1.0f - alpha * alpha + 1e-6f);
    const float sigi  = 1.0f / (1.0f + expf(-inp));
    g[bt * GI_LD + h]        = alpha;
    g[bt * GI_LD + HIDC + h] = beta * sigi * x;
}

// ---------------------------------------------------------------------------
// Chunked scan: h_t = alpha_t * h_{t-1} + bterm_t ; 8 chunks of 256 per seq.
// ---------------------------------------------------------------------------
#define NCH 8
#define CHL 256

__global__ __launch_bounds__(256)
void rg_scan1(const float* __restrict__ g, float* __restrict__ sumA,
              float* __restrict__ sumB)
{
    const int id = blockIdx.x * blockDim.x + threadIdx.x;
    const int h    = id & (HIDC - 1);
    const int rest = id >> 11;
    const int c    = rest & (NCH - 1);
    const int b    = rest >> 3;
    float Aa = 1.0f, Bv = 0.0f;
    const int t0 = c * CHL;
#pragma unroll 8
    for (int t = t0; t < t0 + CHL; ++t) {
        const long row = (long)b * TC + t;
        const float a  = g[row * GI_LD + h];
        const float bt = g[row * GI_LD + HIDC + h];
        Aa *= a;
        Bv = a * Bv + bt;
    }
    const int s = b * HIDC + h;
    sumA[s * NCH + c] = Aa;
    sumB[s * NCH + c] = Bv;
}

__global__ __launch_bounds__(256)
void rg_scan2(const float* __restrict__ sumA, const float* __restrict__ sumB,
              float* __restrict__ carry)
{
    const int s = blockIdx.x * blockDim.x + threadIdx.x;
    float v = 0.0f;
#pragma unroll
    for (int c = 0; c < NCH; ++c) {
        carry[s * NCH + c] = v;
        v = sumA[s * NCH + c] * v + sumB[s * NCH + c];
    }
}

__global__ __launch_bounds__(256)
void rg_scan3(const float* __restrict__ g, const float* __restrict__ carry,
              float* __restrict__ hbuf)
{
    const int id = blockIdx.x * blockDim.x + threadIdx.x;
    const int h    = id & (HIDC - 1);
    const int rest = id >> 11;
    const int c    = rest & (NCH - 1);
    const int b    = rest >> 3;
    float hp = carry[(b * HIDC + h) * NCH + c];
    const int t0 = c * CHL;
#pragma unroll 4
    for (int t = t0; t < t0 + CHL; ++t) {
        const long row = (long)b * TC + t;
        const float a  = g[row * GI_LD + h];
        const float bt = g[row * GI_LD + HIDC + h];
        hp = a * hp + bt;
        hbuf[row * HIDC + h] = hp;
    }
}

// ---------------------------------------------------------------------------
// gelu(gate) * h -> bf16 hi/lo planes (GEMM3's A, compact lda=HID)
// ---------------------------------------------------------------------------
__global__ __launch_bounds__(256)
void rg_gated(const float* __restrict__ gi, const float* __restrict__ hbuf,
              __bf16* __restrict__ ghi, __bf16* __restrict__ glo)
{
    const long idx = (long)blockIdx.x * blockDim.x + threadIdx.x;
    const int  h  = (int)(idx & (HIDC - 1));
    const long bt = idx >> 11;
    const float gate = gi[bt * GI_LD + h];
    const float ge = 0.5f * gate * (1.0f + erff(gate * 0.70710678118f));
    const float v = ge * hbuf[idx];
    const __bf16 hh = (__bf16)v;
    ghi[idx] = hh;
    glo[idx] = (__bf16)(v - (float)hh);
}

// ---------------------------------------------------------------------------
#define MB(x) ((size_t)(x) * 1024 * 1024)

extern "C" void kernel_launch(void* const* d_in, const int* in_sizes, int n_in,
                              void* d_out, int out_size, void* d_ws, size_t ws_size,
                              hipStream_t stream)
{
    (void)in_sizes; (void)n_in; (void)out_size; (void)ws_size;

    const float* x      = (const float*)d_in[0];
    const float* W_in   = (const float*)d_in[1];
    const float* conv_w = (const float*)d_in[2];
    const float* conv_b = (const float*)d_in[3];
    const float* W_g    = (const float*)d_in[4];
    const float* b_g    = (const float*)d_in[5];
    const float* fbase  = (const float*)d_in[6];
    const float* W_out  = (const float*)d_in[7];
    float* out = (float*)d_out;

    char* ws = (char*)d_ws;
    float*  gi   = (float*)(ws);                 // [8192,4096] f32  128MB
    float*  g    = (float*)(ws + MB(128));       // [8192,4096] f32  128MB
    float*  xcv  = (float*)(ws + MB(256));       // [8192,2048] f32   64MB (later h)
    __bf16* ach  = (__bf16*)(ws + MB(320));      // act planes (xc, then gated) 64MB
    __bf16* acl  = ach + (size_t)NROW * HIDC;
    __bf16* xh   = (__bf16*)(ws + MB(384));      // x planes 32MB
    __bf16* xl   = xh + (size_t)NROW * DIMC;
    __bf16* wih  = (__bf16*)(ws + MB(416));      // W_in planes 16MB
    __bf16* wil  = wih + (size_t)(2 * HIDC) * DIMC;
    __bf16* wgh  = (__bf16*)(ws + MB(432));      // W_g planes 32MB
    __bf16* wgl  = wgh + (size_t)(2 * HIDC) * HIDC;
    __bf16* woh  = (__bf16*)(ws + MB(464));      // W_out planes 8MB
    __bf16* wol  = woh + (size_t)DIMC * HIDC;
    float*  sumA = (float*)(ws + MB(472));
    float*  sumB = sumA + 65536;
    float*  carry = sumB + 65536;

    const dim3 blk(256);
    const int ewGrid = (int)(((long)NROW * HIDC) / 256);

    // 0) split GEMM operands into bf16 hi/lo planes (convert-once)
    rg_split<<<(NROW * DIMC / 4) / 256, blk, 0, stream>>>(x, xh, xl);
    rg_split<<<(2 * HIDC * DIMC / 4) / 256, blk, 0, stream>>>(W_in, wih, wil);
    rg_split<<<(2 * HIDC * HIDC / 4) / 256, blk, 0, stream>>>(W_g, wgh, wgl);
    rg_split<<<(DIMC * HIDC / 4) / 256, blk, 0, stream>>>(W_out, woh, wol);

    // 1) gi = x @ W_in^T
    rg_gemm_bf16x3<<<dim3(2 * HIDC / TILE_N, NROW / TILE_M), blk, 0, stream>>>(
        xh, xl, wih, wil, gi, nullptr, DIMC, DIMC, GI_LD);

    // 2) causal depthwise conv (+ bf16 planes for GEMM2)
    rg_dwconv<<<ewGrid, blk, 0, stream>>>(gi, conv_w, conv_b, xcv, ach, acl);

    // 3) g = xcv @ W_g^T + b_g
    rg_gemm_bf16x3<<<dim3(2 * HIDC / TILE_N, NROW / TILE_M), blk, 0, stream>>>(
        ach, acl, wgh, wgl, g, b_g, HIDC, HIDC, GI_LD);

    // 4) gates: g -> (alpha, bterm) in place
    rg_gates<<<ewGrid, blk, 0, stream>>>(g, xcv, fbase);

    // 5) chunked linear-recurrence scan -> h into xcv
    rg_scan1<<<(BC * HIDC * NCH) / 256, blk, 0, stream>>>(g, sumA, sumB);
    rg_scan2<<<(BC * HIDC) / 256, blk, 0, stream>>>(sumA, sumB, carry);
    rg_scan3<<<(BC * HIDC * NCH) / 256, blk, 0, stream>>>(g, carry, xcv);

    // 6) gelu(gate)*h -> bf16 planes (reuse act-plane region)
    rg_gated<<<ewGrid, blk, 0, stream>>>(gi, xcv, ach, acl);

    // 7) out = gated @ W_out^T
    rg_gemm_bf16x3<<<dim3(DIMC / TILE_N, NROW / TILE_M), blk, 0, stream>>>(
        ach, acl, woh, wol, out, nullptr, HIDC, HIDC, DIMC);
}
